// AttnCalc_38628935860291
// MI455X (gfx1250) — compile-verified, matrix operand order checked by reading, not run on hardware
//
#include <hip/hip_runtime.h>
#include <hip/hip_bf16.h>
#include <math.h>

#define Bdim 64
#define Ldim 512
#define Hdim 512

typedef __attribute__((ext_vector_type(16))) __bf16 v16bf;
typedef __attribute__((ext_vector_type(16))) short  v16s;
typedef __attribute__((ext_vector_type(8)))  float  v8f;

__device__ __forceinline__ unsigned short f2bf(float f) {
    unsigned int u = __float_as_uint(f);
    u += 0x7FFFu + ((u >> 16) & 1u);   // round-to-nearest-even truncation
    return (unsigned short)(u >> 16);
}

// ---------------------------------------------------------------------------
// Kernel 1: gather cvg_w[:, :, 0, center] slice into contiguous weff, and
// pre-pack attn_w into bf16 laid out exactly as the WMMA B operand:
//   element i of lane's v16bf  ->  k = (lane>>4)*8 + (i<8 ? i : i+8)
// so the hot loop issues two contiguous b128 loads per lane per K-step.
// ---------------------------------------------------------------------------
__global__ __launch_bounds__(256)
void prep_kernel(const float* __restrict__ cvg_w, const float* __restrict__ attn_w,
                 float* __restrict__ weff, unsigned short* __restrict__ Bpack) {
    int idx = blockIdx.x * blockDim.x + threadIdx.x;   // 0 .. 524287
    if (idx < Ldim * Ldim) {
        int l = idx >> 9, j = idx & 511;
        weff[idx] = cvg_w[((size_t)l * Ldim + j) * (size_t)Hdim + (Hdim - 1) / 2];
    } else {
        int p    = idx - Ldim * Ldim;                  // packed element id
        int i    = p & 15;
        int lane = (p >> 4) & 31;
        int kc   = (p >> 9) & 15;
        int ot   = p >> 13;
        int o    = ot * 16 + (lane & 15);
        int h    = kc * 32 + (lane >> 4) * 8 + (i < 8 ? i : i + 8);
        Bpack[p] = f2bf(attn_w[(size_t)o * Hdim + h]);
    }
}

// ---------------------------------------------------------------------------
// Kernel 2: dec_feat[b,o] and cov_feat[b,l] — one wave per output element,
// coalesced streaming + shuffle reduction (tiny: ~67 MFLOP total).
// ---------------------------------------------------------------------------
__global__ __launch_bounds__(256)
void small_feats_kernel(const float* __restrict__ hidden, const float* __restrict__ dec_w,
                        const float* __restrict__ dec_b,  const float* __restrict__ coverage,
                        const float* __restrict__ weff,   const float* __restrict__ cvg_b,
                        float* __restrict__ dec_feat,     float* __restrict__ cov_feat) {
    int g    = blockIdx.x * blockDim.x + threadIdx.x;
    int wv   = g >> 5;
    int lane = g & 31;
    if (wv < Bdim * Hdim) {
        int b = wv >> 9, o = wv & 511;
        float acc = 0.0f;
        for (int j = lane; j < Hdim; j += 32)
            acc += hidden[b * Hdim + j] * dec_w[(size_t)o * Hdim + j];
        for (int off = 16; off >= 1; off >>= 1) acc += __shfl_xor(acc, off, 32);
        if (lane == 0) dec_feat[b * Hdim + o] = acc + dec_b[o];
    } else {
        int w2 = wv - Bdim * Hdim;
        int b = w2 >> 9, l = w2 & 511;
        float acc = 0.0f;
        for (int j = lane; j < Ldim; j += 32)
            acc += coverage[b * Ldim + j] * weff[(size_t)l * Ldim + j];
        for (int off = 16; off >= 1; off >>= 1) acc += __shfl_xor(acc, off, 32);
        if (lane == 0) cov_feat[b * Ldim + l] = acc + cvg_b[l];
    }
}

// ---------------------------------------------------------------------------
// Kernel 3: fused enc_feat GEMM (bf16 WMMA) + tanh + dot-with-v -> scores.
// One wave owns a 16-row L tile, keeps the 16x512 bf16 A tile in VGPRs
// (16 x v16bf = 128 VGPRs) and sweeps all 32 o-tiles: 512 WMMAs per wave.
// feats (B*L*H, 67 MB) is never materialized.
// ---------------------------------------------------------------------------
__global__ __launch_bounds__(128)
void attn_wmma_kernel(const float* __restrict__ enc,
                      const unsigned short* __restrict__ Bpack,
                      const float* __restrict__ attn_b,
                      const float* __restrict__ dec_feat,
                      const float* __restrict__ cov_feat,
                      const float* __restrict__ v,
                      float* __restrict__ scores) {
    const int lane  = threadIdx.x & 31;
    const int wave  = threadIdx.x >> 5;
    const int b     = blockIdx.x >> 3;         // L/64 = 8 blocks per batch
    const int lblk  = blockIdx.x & 7;
    const int lbase = lblk * 64 + wave * 16;
    const int row   = lane & 15;
    const int hhalf = lane >> 4;               // 0 or 1

    // --- load & pack A (encoder rows) in WMMA A layout, fp32 -> bf16 ---
    v16bf A[16];
    const float* encRow = enc + ((size_t)b * Ldim + lbase + row) * Hdim;
    #pragma unroll
    for (int c = 0; c < 16; ++c) {
        const int kb = c * 32 + hhalf * 8;
        const float4 f0 = *reinterpret_cast<const float4*>(encRow + kb);
        const float4 f1 = *reinterpret_cast<const float4*>(encRow + kb + 4);
        const float4 f2 = *reinterpret_cast<const float4*>(encRow + kb + 16);
        const float4 f3 = *reinterpret_cast<const float4*>(encRow + kb + 20);
        v16s t;
        t[0]  = (short)f2bf(f0.x); t[1]  = (short)f2bf(f0.y);
        t[2]  = (short)f2bf(f0.z); t[3]  = (short)f2bf(f0.w);
        t[4]  = (short)f2bf(f1.x); t[5]  = (short)f2bf(f1.y);
        t[6]  = (short)f2bf(f1.z); t[7]  = (short)f2bf(f1.w);
        t[8]  = (short)f2bf(f2.x); t[9]  = (short)f2bf(f2.y);
        t[10] = (short)f2bf(f2.z); t[11] = (short)f2bf(f2.w);
        t[12] = (short)f2bf(f3.x); t[13] = (short)f2bf(f3.y);
        t[14] = (short)f2bf(f3.z); t[15] = (short)f2bf(f3.w);
        A[c] = __builtin_bit_cast(v16bf, t);
    }

    // per-row coverage term (row m = hhalf*8 + j)
    float covf[8];
    #pragma unroll
    for (int j = 0; j < 8; ++j)
        covf[j] = cov_feat[b * Ldim + lbase + hhalf * 8 + j];

    float sAcc[8] = {0.f, 0.f, 0.f, 0.f, 0.f, 0.f, 0.f, 0.f};

    for (int ot = 0; ot < 32; ++ot) {
        const int   o    = ot * 16 + (lane & 15);
        const float addo = attn_b[o] + dec_feat[b * Hdim + o];
        const float vv   = v[b * Hdim + o];

        v8f acc = {0.f, 0.f, 0.f, 0.f, 0.f, 0.f, 0.f, 0.f};
        const v16s* bp = reinterpret_cast<const v16s*>(Bpack) +
                         ((size_t)ot * 16 * 32 + lane);
        #pragma unroll
        for (int kc = 0; kc < 16; ++kc) {
            v16bf Bt = __builtin_bit_cast(v16bf, bp[kc * 32]);
            acc = __builtin_amdgcn_wmma_f32_16x16x32_bf16(
                      false, A[kc], false, Bt, (short)0, acc, false, false);
        }
        #pragma unroll
        for (int j = 0; j < 8; ++j) {
            const float f = tanhf(acc[j] + addo + covf[j]);
            sAcc[j] += f * vv;   // partial reduce over the 16 o's of this tile
        }
    }

    // reduce over the 16 columns held by each half-wave (xor 1,2,4,8 stays in-half)
    #pragma unroll
    for (int j = 0; j < 8; ++j) {
        float s = sAcc[j];
        s += __shfl_xor(s, 1, 32);
        s += __shfl_xor(s, 2, 32);
        s += __shfl_xor(s, 4, 32);
        s += __shfl_xor(s, 8, 32);
        if ((lane & 15) == 0)
            scores[b * Ldim + lbase + hhalf * 8 + j] = s;
    }
}

// ---------------------------------------------------------------------------
// Kernel 4: masked softmax over L, attn_weights + new_coverage outputs,
// then context[b,h] = sum_l w[l]*enc[b,l,h] (coalesced, enc is L2-resident).
// ---------------------------------------------------------------------------
__global__ __launch_bounds__(256)
void softmax_ctx_kernel(const float* __restrict__ scores,
                        const long long* __restrict__ mask,
                        const float* __restrict__ coverage,
                        const float* __restrict__ enc,
                        float* __restrict__ out) {
    const int b   = blockIdx.x;
    const int tid = threadIdx.x;
    __shared__ float w[Ldim];
    __shared__ float red[256];

    float s0 = scores[b * Ldim + tid];
    float s1 = scores[b * Ldim + tid + 256];
    if (mask[b * Ldim + tid] != 1)       s0 = -INFINITY;
    if (mask[b * Ldim + tid + 256] != 1) s1 = -INFINITY;

    red[tid] = fmaxf(s0, s1);
    __syncthreads();
    for (int off = 128; off > 0; off >>= 1) {
        if (tid < off) red[tid] = fmaxf(red[tid], red[tid + off]);
        __syncthreads();
    }
    const float m = red[0];
    __syncthreads();

    const float e0 = expf(s0 - m);
    const float e1 = expf(s1 - m);
    red[tid] = e0 + e1;
    __syncthreads();
    for (int off = 128; off > 0; off >>= 1) {
        if (tid < off) red[tid] += red[tid + off];
        __syncthreads();
    }
    const float inv = 1.0f / red[0];
    __syncthreads();

    const float w0 = e0 * inv, w1 = e1 * inv;
    w[tid] = w0;
    w[tid + 256] = w1;
    // outputs: [context | attn_weights | new_coverage], each Bdim*512 floats
    out[Bdim * Hdim + b * Ldim + tid]            = w0;
    out[Bdim * Hdim + b * Ldim + tid + 256]      = w1;
    out[2 * Bdim * Hdim + b * Ldim + tid]        = coverage[b * Ldim + tid] + w0;
    out[2 * Bdim * Hdim + b * Ldim + tid + 256]  = coverage[b * Ldim + tid + 256] + w1;
    __syncthreads();

    float c0 = 0.f, c1 = 0.f;
    const float* eb = enc + (size_t)b * Ldim * Hdim;
    for (int l = 0; l < Ldim; ++l) {
        const float wl = w[l];
        c0 += wl * eb[(size_t)l * Hdim + tid];
        c1 += wl * eb[(size_t)l * Hdim + tid + 256];
    }
    out[b * Hdim + tid]       = c0;
    out[b * Hdim + tid + 256] = c1;
}

extern "C" void kernel_launch(void* const* d_in, const int* in_sizes, int n_in,
                              void* d_out, int out_size, void* d_ws, size_t ws_size,
                              hipStream_t stream) {
    const float*     enc      = (const float*)d_in[0];
    const long long* mask     = (const long long*)d_in[1];
    const float*     hidden   = (const float*)d_in[2];
    const float*     coverage = (const float*)d_in[3];
    const float*     attn_w   = (const float*)d_in[4];
    const float*     attn_b   = (const float*)d_in[5];
    const float*     dec_w    = (const float*)d_in[6];
    const float*     dec_b    = (const float*)d_in[7];
    const float*     cvg_w    = (const float*)d_in[8];
    const float*     cvg_b    = (const float*)d_in[9];
    const float*     v        = (const float*)d_in[10];
    float* out = (float*)d_out;

    char* ws = (char*)d_ws;
    float*          weff     = (float*)ws;                                   // 1 MB
    unsigned short* Bpack    = (unsigned short*)(ws + (1 << 20));            // 512 KB
    float*          dec_feat = (float*)(ws + (1 << 20) + (1 << 19));         // 128 KB
    float*          cov_feat = dec_feat + Bdim * Hdim;                       // 128 KB
    float*          scores   = cov_feat + Bdim * Ldim;                       // 128 KB

    prep_kernel<<<2048, 256, 0, stream>>>(cvg_w, attn_w, weff, Bpack);
    small_feats_kernel<<<8192, 256, 0, stream>>>(hidden, dec_w, dec_b, coverage,
                                                 weff, cvg_b, dec_feat, cov_feat);
    attn_wmma_kernel<<<Bdim * (Ldim / 64), 128, 0, stream>>>(enc, Bpack, attn_b,
                                                             dec_feat, cov_feat, v, scores);
    softmax_ctx_kernel<<<Bdim, 256, 0, stream>>>(scores, mask, coverage, enc, out);
}
